// RescostRefinement_66348654789212
// MI455X (gfx1250) — compile-verified, matrix operand order checked by reading, not run on hardware
//
#include <hip/hip_runtime.h>
#include <hip/hip_bf16.h>

typedef _Float16 h16;
typedef __attribute__((ext_vector_type(16))) _Float16 v16h;
typedef __attribute__((ext_vector_type(8)))  float    v8f;

#define NB 4
#define HH 384
#define WW 768
#define HW (HH*WW)
#define PP (NB*HW)          // 1179648 pixels
#define BN_EPS 1e-5f

// ---------------- weight packing: OIHW f32 -> [CoutPad][Kpad] f16 --------------
__global__ void pack_w_k(const float* __restrict__ w, h16* __restrict__ out,
                         int Cout, int K, int CoutPad, int Kpad) {
  int i = blockIdx.x * 256 + threadIdx.x;
  int total = CoutPad * Kpad;
  if (i >= total) return;
  int row = i / Kpad, k = i - row * Kpad;
  float v = (row < Cout && k < K) ? w[(size_t)row * K + k] : 0.f;
  out[i] = (h16)v;
}

// ---------------- implicit-GEMM conv with WMMA f16 -> f32 ----------------------
// A: packed weights [CoutPad][Kpad] f16 (Kpad multiple of 64).
// X: activations [Cin][PP] f16.  Out: raw conv output [CoutReal][PP] f32.
// Template: CT = CoutPad/16 (1,2,4), KS = kernel size (1,3).
// Each wave computes a 16(cout) x 32(pixel) tile: the A fragment is loaded once
// per 32-K group and reused for two WMMAs (two 16-pixel B subtiles).
template <int CT, int KS>
__global__ void __launch_bounds__(256) conv_wmma_k(
    const h16* __restrict__ A, const h16* __restrict__ X, float* __restrict__ Out,
    int Cin, int CoutReal, int Kpad, int dil) {
  constexpr int CoutPad   = CT * 16;
  constexpr int PW        = 8 / CT;       // waves along pixel dim
  constexpr int pixPerBlk = PW * 32;      // 32 pixels per wave
  constexpr int KK        = KS * KS;

  __shared__ h16 As[CoutPad * 64];        // [CoutPad][64]
  __shared__ h16 Bs[pixPerBlk * 64];      // [pixPerBlk][64]

  const int lane     = threadIdx.x & 31;
  const int wave     = threadIdx.x >> 5;
  const int coutTile = wave % CT;
  const int pixTile  = wave / CT;         // 0..PW-1
  const int pixBase  = blockIdx.x * pixPerBlk;
  const int Kreal    = Cin * KK;
  const int padw     = dil * (KS >> 1);

  // Fixed per-thread K element for B staging (256 threads, 64-wide K tile).
  const int klB    = threadIdx.x & 63;
  const int plBase = threadIdx.x >> 6;    // 0..3

  v8f acc0 = {}, acc1 = {};

  for (int kk = 0; kk < Kpad; kk += 64) {
    // prefetch next weight K-slab (global_prefetch_b8)
    if (kk + 64 < Kpad && threadIdx.x < (unsigned)CoutPad)
      __builtin_prefetch(A + (size_t)threadIdx.x * Kpad + kk + 64, 0, 0);

    __syncthreads();
    // ---- stage A tile: CoutPad*64 f16 = CoutPad*8 uint4, coalesced ----
    for (int i = threadIdx.x; i < CoutPad * 8; i += 256) {
      int row = i >> 3, q = i & 7;
      ((uint4*)As)[i] = *((const uint4*)(A + (size_t)row * Kpad + kk) + q);
    }
    // ---- stage B tile (im2row gather): k-invariant work hoisted ----
    {
      int k = kk + klB;
      bool kvalid = (k < Kreal);
      int c = k / KK;
      int tap = k - c * KK;
      int ty = (KS == 1) ? 0 : tap / KS;
      int tx = (KS == 1) ? 0 : tap - ty * KS;
      int oy = ty * dil - padw;
      int ox = tx * dil - padw;
      const h16* Xc = X + (size_t)c * PP;
#pragma unroll 4
      for (int pl = plBase; pl < pixPerBlk; pl += 4) {
        int p = pixBase + pl;
        h16 v = (h16)0.f;
        if (p < PP && kvalid) {
          int rem = p % HW;
          int y = rem / WW, x = rem - y * WW;
          int ys = y + oy, xs = x + ox;
          if ((unsigned)ys < (unsigned)HH && (unsigned)xs < (unsigned)WW)
            v = Xc[(p - rem) + ys * WW + xs];
        }
        Bs[pl * 64 + klB] = v;
      }
    }
    __syncthreads();

    // ---- 4 x WMMA over the 64-wide K tile (A fragment reused twice) ----
    // 16-bit A/B 16x32 striping per 32-K group g:
    // lanes 0-15 take uint4 {4g, 4g+2}; lanes 16-31 take {4g+1, 4g+3}.
    const int lh = lane >> 4;
    const uint4* Ar  = (const uint4*)(As + (coutTile * 16 + (lane & 15)) * 64);
    const uint4* Br0 = (const uint4*)(Bs + (pixTile * 32 + (lane & 15)) * 64);
    const uint4* Br1 = Br0 + 16 * 8;     // +16 rows of 64 f16 (8 uint4 each)
    union FR { uint4 q[2]; v16h h; } fa, fb0, fb1;
#pragma unroll
    for (int g = 0; g < 2; g++) {
      fa.q[0]  = Ar[4 * g + lh];  fa.q[1]  = Ar[4 * g + lh + 2];
      fb0.q[0] = Br0[4 * g + lh]; fb0.q[1] = Br0[4 * g + lh + 2];
      fb1.q[0] = Br1[4 * g + lh]; fb1.q[1] = Br1[4 * g + lh + 2];
      acc0 = __builtin_amdgcn_wmma_f32_16x16x32_f16(
          false, fa.h, false, fb0.h, (short)0, acc0, false, false);
      acc1 = __builtin_amdgcn_wmma_f32_16x16x32_f16(
          false, fa.h, false, fb1.h, (short)0, acc1, false, false);
    }
  }

  // C/D layout: VGPR r -> M = r (+8 for lanes 16-31), N = lane&15
  const int mbase = coutTile * 16 + ((lane >= 16) ? 8 : 0);
  int n0 = pixBase + pixTile * 32 + (lane & 15);
  int n1 = n0 + 16;
  if (n0 < PP) {
#pragma unroll
    for (int r = 0; r < 8; r++) {
      int m = mbase + r;
      if (m < CoutReal) Out[(size_t)m * PP + n0] = acc0[r];
    }
  }
  if (n1 < PP) {
#pragma unroll
    for (int r = 0; r < 8; r++) {
      int m = mbase + r;
      if (m < CoutReal) Out[(size_t)m * PP + n1] = acc1[r];
    }
  }
}

// ---------------- BN statistics (sum, sumsq per channel) -----------------------
__global__ void zero_k(float* p, int n) {
  int i = blockIdx.x * 256 + threadIdx.x;
  if (i < n) p[i] = 0.f;
}

__global__ void stats_k(const float* __restrict__ X, float* __restrict__ st) {
  int c = blockIdx.y;
  float s = 0.f, s2 = 0.f;
  for (long long i = (long long)blockIdx.x * 256 + threadIdx.x; i < PP;
       i += (long long)gridDim.x * 256) {
    float v = X[(size_t)c * PP + i];
    s += v; s2 += v * v;
  }
  __shared__ float sh[512];
  sh[threadIdx.x] = s; sh[256 + threadIdx.x] = s2;
  __syncthreads();
  for (int o = 128; o; o >>= 1) {
    if (threadIdx.x < (unsigned)o) {
      sh[threadIdx.x]       += sh[threadIdx.x + o];
      sh[256 + threadIdx.x] += sh[256 + threadIdx.x + o];
    }
    __syncthreads();
  }
  if (threadIdx.x == 0) {
    atomicAdd(&st[2 * c],     sh[0]);
    atomicAdd(&st[2 * c + 1], sh[256]);
  }
}

// ---------------- BN apply + optional residual + activation -> f16 -------------
// act: 0 = none, 1 = relu, 2 = leaky(0.2).  Grid: (ceil(PP/256), C).
__global__ void bn_apply_k(const float* __restrict__ raw, const float* __restrict__ st,
                           const float* __restrict__ g, const float* __restrict__ bb,
                           const h16* __restrict__ resid, h16* __restrict__ out,
                           int act) {
  int p = blockIdx.x * 256 + threadIdx.x;
  if (p >= PP) return;
  int c = blockIdx.y;
  size_t i = (size_t)c * PP + p;
  float mean = st[2 * c] * (1.f / PP);
  float var  = st[2 * c + 1] * (1.f / PP) - mean * mean;
  float sc = g[c] * rsqrtf(var + BN_EPS);
  float sh = bb[c] - mean * sc;
  float v = raw[i] * sc + sh;
  if (resid) v += (float)resid[i];
  if (act == 1) v = fmaxf(v, 0.f);
  else if (act == 2) v = (v > 0.f) ? v : 0.2f * v;
  out[i] = (h16)v;
}

// ---------------- NCHW f32 image -> [C][PP] f16 --------------------------------
__global__ void img_h_k(const float* __restrict__ img, h16* __restrict__ out) {
  int i = blockIdx.x * 256 + threadIdx.x;
  if (i >= 3 * PP) return;
  int c = i / PP, p = i - c * PP;
  int n = p / HW, rem = p - n * HW;
  out[i] = (h16)img[((size_t)n * 3 + c) * HW + rem];
}

// ---------------- disparity 2x bilinear upsample * scale -----------------------
__global__ void upsample_k(const float* __restrict__ low, float* __restrict__ disp,
                           h16* __restrict__ o1, h16* __restrict__ o2) {
  int i = blockIdx.x * 256 + threadIdx.x;
  if (i >= PP) return;
  int n = i / HW, rem = i - n * HW, y = rem / WW, x = rem - y * WW;
  const int Hs = HH / 2, Ws = WW / 2;
  float sy = (y + 0.5f) * 0.5f - 0.5f, sx = (x + 0.5f) * 0.5f - 0.5f;
  float fy = sy - floorf(sy), fx = sx - floorf(sx);
  int y0 = (int)floorf(sy), x0 = (int)floorf(sx);
  int y0c = min(max(y0, 0), Hs - 1), y1c = min(max(y0 + 1, 0), Hs - 1);
  int x0c = min(max(x0, 0), Ws - 1), x1c = min(max(x0 + 1, 0), Ws - 1);
  const float* L = low + (size_t)n * Hs * Ws;
  float v = (L[y0c * Ws + x0c] * (1.f - fx) + L[y0c * Ws + x1c] * fx) * (1.f - fy)
          + (L[y1c * Ws + x0c] * (1.f - fx) + L[y1c * Ws + x1c] * fx) * fy;
  v *= 2.0f;  // scale = W / (W/2)
  disp[i] = v; o1[i] = (h16)v; o2[i] = (h16)v;
}

// ---------------- warped right + |err| -----------------------------------------
__global__ void warp_err_k(const float* __restrict__ disp, const h16* __restrict__ rimg,
                           const h16* __restrict__ limg, h16* __restrict__ occ,
                           h16* __restrict__ att) {
  int p = blockIdx.x * 256 + threadIdx.x;
  if (p >= PP) return;
  int rem = p % HW; int x = rem % WW; int rowbase = p - x;
  float xs = (float)x - disp[p];
  float x0f = floorf(xs); float fx = xs - x0f; int xi = (int)x0f;
  bool v0 = (xi >= 0) && (xi < WW);
  bool v1 = (xi + 1 >= 0) && (xi + 1 < WW);
  int xc0 = min(max(xi, 0), WW - 1), xc1 = min(max(xi + 1, 0), WW - 1);
#pragma unroll
  for (int c = 0; c < 3; c++) {
    float g0 = v0 ? (float)rimg[(size_t)c * PP + rowbase + xc0] : 0.f;
    float g1 = v1 ? (float)rimg[(size_t)c * PP + rowbase + xc1] : 0.f;
    float w = g0 * (1.f - fx) + g1 * fx;
    float e = fabsf(w - (float)limg[(size_t)c * PP + p]);
    occ[(size_t)c * PP + p] = (h16)e;
    att[(size_t)c * PP + p] = (h16)e;
  }
}

// ---------------- 9-hypothesis correlation volume ------------------------------
__global__ void corr_k(const float* __restrict__ disp, const h16* __restrict__ lf,
                       const h16* __restrict__ rf, h16* __restrict__ outv) {
  int p = blockIdx.x * 256 + threadIdx.x;
  if (p >= PP) return;
  int rem = p % HW; int x = rem % WW; int rowbase = p - x;
  float d0 = disp[p];
  for (int i = 0; i < 9; i++) {
    float xs = (float)x + (float)(i - 4) - d0;
    float x0f = floorf(xs); float fx = xs - x0f; int xi = (int)x0f;
    bool v0 = (xi >= 0) && (xi < WW);
    bool v1 = (xi + 1 >= 0) && (xi + 1 < WW);
    int xc0 = min(max(xi, 0), WW - 1), xc1 = min(max(xi + 1, 0), WW - 1);
    float acc = 0.f;
#pragma unroll 8
    for (int c = 0; c < 32; c++) {
      float g0 = v0 ? (float)rf[(size_t)c * PP + rowbase + xc0] : 0.f;
      float g1 = v1 ? (float)rf[(size_t)c * PP + rowbase + xc1] : 0.f;
      acc += (float)lf[(size_t)c * PP + p] * (g0 * (1.f - fx) + g1 * fx);
    }
    outv[(size_t)i * PP + p] = (h16)(acc * (1.f / 32.f));
  }
}

// ---------------- sigmoid of raw conv -> two f16 destinations -------------------
__global__ void sigmoid_k(const float* __restrict__ raw, h16* d1, h16* d2) {
  int p = blockIdx.x * 256 + threadIdx.x;
  if (p >= PP) return;
  float v = 1.f / (1.f + __expf(-raw[p]));
  d1[p] = (h16)v; d2[p] = (h16)v;
}

// ---------------- f16 elementwise add: b += a ----------------------------------
__global__ void add_h_k(const h16* __restrict__ a, h16* __restrict__ b, long long n) {
  long long i = (long long)blockIdx.x * 256 + threadIdx.x;
  if (i < n) b[i] = (h16)((float)a[i] + (float)b[i]);
}

// ---------------- final residual combine ---------------------------------------
__global__ void final_k(const float* __restrict__ disp, const float* __restrict__ satt,
                        const float* __restrict__ ores, const float* __restrict__ eres,
                        float* __restrict__ out) {
  int p = blockIdx.x * 256 + threadIdx.x;
  if (p >= PP) return;
  float a = 1.f / (1.f + __expf(-satt[p]));
  float r = a * ores[p] + (1.f - a) * eres[p];
  out[p] = fmaxf(disp[p] + r, 0.f);
}

// ================================================================================
extern "C" void kernel_launch(void* const* d_in, const int* in_sizes, int n_in,
                              void* d_out, int out_size, void* d_ws, size_t ws_size,
                              hipStream_t stream) {
  (void)in_sizes; (void)n_in; (void)out_size; (void)ws_size;
  const float* low   = (const float*)d_in[0];
  const float* left  = (const float*)d_in[1];
  const float* right = (const float*)d_in[2];
  auto W_ = [&](int i) { return (const float*)d_in[i]; };

  // ---- workspace carve ----
  char* ws = (char*)d_ws;
  size_t cur = 0;
  auto alloc = [&](size_t bytes) -> char* {
    char* r = ws + cur;
    cur = (cur + bytes + 255) & ~(size_t)255;
    return r;
  };
  float* b_raw    = (float*)alloc(sizeof(float) * 64ull * PP);
  h16*   b_lf     = (h16*)alloc(2ull * 32 * PP);
  h16*   b_rf     = (h16*)alloc(2ull * 32 * PP);
  h16*   b_occin  = (h16*)alloc(2ull * 7 * PP);   // err(3), right(3), disp(1)
  h16*   b_attin  = (h16*)alloc(2ull * 8 * PP);   // err(3), left(3), disp(1), mask(1)
  h16*   b_oft    = (h16*)alloc(2ull * 10 * PP);  // res_volume(9), mask(1)
  h16*   b_x0     = (h16*)alloc(2ull * 64 * PP);
  h16*   b_x1     = (h16*)alloc(2ull * 64 * PP);
  h16*   b_res    = (h16*)alloc(2ull * 64 * PP);
  float* b_disp   = (float*)alloc(sizeof(float) * PP);
  float* b_praw   = (float*)alloc(sizeof(float) * PP);
  float* b_attraw = (float*)alloc(sizeof(float) * PP);
  float* b_occres = (float*)alloc(sizeof(float) * PP);
  float* b_errres = (float*)alloc(sizeof(float) * PP);
  float* b_stats  = (float*)alloc(sizeof(float) * 256);
  h16*   wslab    = (h16*)alloc(2ull * 4 * 1024 * 1024);
  h16*   wcur     = wslab;

  // ---- launch helpers ----
  auto pack = [&](int widx, int Cout, int Cin, int ks, int CoutPad) -> h16* {
    int K = Cin * ks * ks, Kpad = (K + 63) & ~63;
    h16* dst = wcur;
    wcur += (size_t)CoutPad * Kpad;
    int total = CoutPad * Kpad;
    pack_w_k<<<dim3((total + 255) / 256), dim3(256), 0, stream>>>(
        W_(widx), dst, Cout, K, CoutPad, Kpad);
    return dst;
  };
  auto conv = [&](const h16* wp, const h16* X, float* out, int Cin, int CoutPad,
                  int CoutReal, int ks, int dil) {
    int Kpad = ((Cin * ks * ks) + 63) & ~63;
    int CT = CoutPad >> 4;
    int pixPerBlk = (8 / CT) * 32;
    dim3 g((PP + pixPerBlk - 1) / pixPerBlk);
    if (CT == 1 && ks == 3)
      conv_wmma_k<1, 3><<<g, dim3(256), 0, stream>>>(wp, X, out, Cin, CoutReal, Kpad, dil);
    else if (CT == 1 && ks == 1)
      conv_wmma_k<1, 1><<<g, dim3(256), 0, stream>>>(wp, X, out, Cin, CoutReal, Kpad, dil);
    else if (CT == 2 && ks == 3)
      conv_wmma_k<2, 3><<<g, dim3(256), 0, stream>>>(wp, X, out, Cin, CoutReal, Kpad, dil);
    else if (CT == 2 && ks == 1)
      conv_wmma_k<2, 1><<<g, dim3(256), 0, stream>>>(wp, X, out, Cin, CoutReal, Kpad, dil);
    else if (CT == 4 && ks == 3)
      conv_wmma_k<4, 3><<<g, dim3(256), 0, stream>>>(wp, X, out, Cin, CoutReal, Kpad, dil);
    else
      conv_wmma_k<4, 1><<<g, dim3(256), 0, stream>>>(wp, X, out, Cin, CoutReal, Kpad, dil);
  };
  auto bnapply = [&](int C, int gi, int bi, const h16* resid, h16* out, int act) {
    zero_k<<<dim3(1), dim3(256), 0, stream>>>(b_stats, 2 * C);
    stats_k<<<dim3(64, C), dim3(256), 0, stream>>>(b_raw, b_stats);
    bn_apply_k<<<dim3((PP + 255) / 256, C), dim3(256), 0, stream>>>(
        b_raw, b_stats, W_(gi), W_(bi), resid, out, act);
  };
  auto cbr = [&](int widx, int gi, int bi, int Cout, int Cin, int ks, int dil,
                 const h16* X, h16* out, int act, const h16* resid) {
    int CoutPad = (Cout + 15) & ~15;
    h16* wp = pack(widx, Cout, Cin, ks, CoutPad);
    conv(wp, X, b_raw, Cin, CoutPad, Cout, ks, dil);
    bnapply(Cout, gi, bi, resid, out, act);
  };

  const unsigned gP = (PP + 255) / 256;
  h16* rightH = b_occin + 3ull * PP;  // right image f16 lives in occ_in ch3-5
  h16* leftH  = b_attin + 3ull * PP;  // left image f16 lives in att_in ch3-5

  // ---- preprocessing ----
  img_h_k<<<dim3((3 * PP + 255) / 256), dim3(256), 0, stream>>>(left, leftH);
  img_h_k<<<dim3((3 * PP + 255) / 256), dim3(256), 0, stream>>>(right, rightH);
  upsample_k<<<dim3(gP), dim3(256), 0, stream>>>(low, b_disp,
                                                 b_occin + 6ull * PP,
                                                 b_attin + 6ull * PP);

  // ---- feature extractor (shared weights, leaky 0.2) ----
  {
    h16* wf = pack(3, 32, 3, 3, 32);
    conv(wf, leftH, b_raw, 3, 32, 32, 3, 1);
    bnapply(32, 4, 5, nullptr, b_lf, 2);
    conv(wf, rightH, b_raw, 3, 32, 32, 3, 1);
    bnapply(32, 4, 5, nullptr, b_rf, 2);
  }

  // ---- warped error + correlation volume ----
  warp_err_k<<<dim3(gP), dim3(256), 0, stream>>>(b_disp, rightH, leftH,
                                                 b_occin, b_attin);
  corr_k<<<dim3(gP), dim3(256), 0, stream>>>(b_disp, b_lf, b_rf, b_oft);

  // ---- occlusion-prediction trunk ----
  cbr(6, 7, 8, 32, 7, 3, 1, b_occin, b_x0, 1, nullptr);        // op_proj1
  cbr(9, 10, 11, 32, 32, 1, 1, b_x0, b_x1, 0, nullptr);        // op_proj2 (identity)
  // op block 0 (dil 2)
  cbr(12, 13, 14, 32, 32, 1, 1, b_x1, b_x0, 1, nullptr);
  cbr(15, 16, 17, 32, 32, 3, 2, b_x0, b_x0, 1, nullptr);
  cbr(18, 19, 20, 32, 32, 1, 1, b_x0, b_x0, 1, nullptr);
  add_h_k<<<dim3((unsigned)((32ll * PP + 255) / 256)), dim3(256), 0, stream>>>(
      b_x0, b_x1, 32ll * PP);
  // op block 1 (dil 4)
  cbr(21, 22, 23, 32, 32, 1, 1, b_x1, b_x0, 1, nullptr);
  cbr(24, 25, 26, 32, 32, 3, 4, b_x0, b_x0, 1, nullptr);
  cbr(27, 28, 29, 32, 32, 1, 1, b_x0, b_x0, 1, nullptr);
  add_h_k<<<dim3((unsigned)((32ll * PP + 255) / 256)), dim3(256), 0, stream>>>(
      b_x0, b_x1, 32ll * PP);
  // op_pred 32->1, sigmoid -> mask (occl_feat ch9, att_in ch7)
  {
    h16* wp = pack(30, 1, 32, 3, 16);
    conv(wp, b_x1, b_praw, 32, 16, 1, 3, 1);
    sigmoid_k<<<dim3(gP), dim3(256), 0, stream>>>(b_praw, b_oft + 9ull * PP,
                                                  b_attin + 7ull * PP);
  }

  // ---- spatial attention branch ----
  cbr(31, 32, 33, 16, 8, 1, 1, b_attin, b_x0, 1, nullptr);     // sa1
  cbr(34, 35, 36, 16, 16, 3, 1, b_x0, b_x0, 1, nullptr);       // sa2
  {
    h16* wp = pack(37, 1, 16, 1, 16);
    conv(wp, b_x0, b_attraw, 16, 16, 1, 1, 1);                 // sa3 raw
  }

  // ---- occlusion-residual branch ----
  cbr(38, 39, 40, 32, 10, 3, 1, b_oft, b_x0, 1, nullptr);      // or1
  cbr(41, 42, 43, 64, 32, 3, 1, b_x0, b_x0, 1, nullptr);       // or2
  cbr(44, 45, 46, 64, 64, 3, 1, b_x0, b_x0, 1, nullptr);       // or3
  {
    h16* wp = pack(47, 1, 64, 3, 16);
    conv(wp, b_x0, b_occres, 64, 16, 1, 3, 1);                 // or4 raw
  }

  // ---- aggregation branch ----
  cbr(48, 49, 50, 32, 9, 3, 1, b_oft, b_x0, 1, nullptr);       // agg0
  // rb1 (with shortcut)
  cbr(59, 61, 62, 64, 32, 1, 1, b_x0, b_res, 0, nullptr);      // rb1_sc (no act)
  cbr(51, 53, 54, 64, 32, 3, 1, b_x0, b_x1, 1, nullptr);       // rb1_c1
  cbr(55, 57, 58, 64, 64, 3, 1, b_x1, b_x0, 1, b_res);         // rb1_c2 + res + relu
  // rb2 (identity shortcut)
  cbr(63, 65, 66, 64, 64, 3, 1, b_x0, b_x1, 1, nullptr);       // rb2_c1
  cbr(67, 69, 70, 64, 64, 3, 1, b_x1, b_x1, 1, b_x0);          // rb2_c2 + res + relu
  {
    h16* wp = pack(71, 1, 64, 3, 16);
    conv(wp, b_x1, b_errres, 64, 16, 1, 3, 1);                 // agg_pred raw
  }

  // ---- final combine ----
  final_k<<<dim3(gP), dim3(256), 0, stream>>>(b_disp, b_attraw, b_occres,
                                              b_errres, (float*)d_out);
}
